// FoveatedConv2d_7997229105538
// MI455X (gfx1250) — compile-verified, matrix operand order checked by reading, not run on hardware
//
#include <hip/hip_runtime.h>
#include <hip/hip_bf16.h>

// ---------------------------------------------------------------------------
// FoveatedConv2d fused kernel for gfx1250 (MI455X)
//   B=4, C=16, H=W=192, O=64, K = C*49 = 784 (padded to KP=800 = 25*32)
//   out[b,o,h,w] = sum_k F[b,k,h,w] * weight[o,k] + bias[o]
//   F built on the fly from 661 reflect-padded taps; GEMM via bf16 WMMA.
// ---------------------------------------------------------------------------

typedef __attribute__((ext_vector_type(16))) __bf16 v16bf;
typedef __attribute__((ext_vector_type(8)))  float  v8f;

#define Bdim 4
#define Cdim 16
#define Hdim 192
#define Wdim 192
#define Odim 64
#define Kdim 784           // C * 49
#define KP   800           // padded K (25 chunks of 32)
#define NCHUNK 25
#define TILE_M 32          // pixels per workgroup
#define FSTRIDE 808        // LDS row stride in bf16 elems (16B aligned, conflict-skewed)
#define MAXTAPS 704        // 661 actual taps

// d_ws layout (bytes):
//   [0 .. 102400)            Bpack: bf16 weight, WMMA-B-fragment order
//                            [kchunk(25)][ntile(4)][lane(32)][16 bf16]
//   [102400 .. 103104)       tap di  (int8 x 704)
//   [103104 .. 103808)       tap dj  (int8 x 704)
//   [103808 .. 103936)       pos start (int16 x 64)
//   [103936 .. 104064)       pos len   (int16 x 64)
#define WS_BPACK   0
#define WS_DI      102400
#define WS_DJ      (102400 + 704)
#define WS_START   (102400 + 1408)
#define WS_LEN     (102400 + 1408 + 128)

__device__ __forceinline__ unsigned short f2bf(float f) {
    union { float f; unsigned u; } v; v.f = f;
    unsigned r = v.u + 0x7FFFu + ((v.u >> 16) & 1u);   // round-to-nearest-even
    return (unsigned short)(r >> 16);
}

__device__ __forceinline__ int refl192(int t) {
    // clip to padded range [-3, 194] then reflect (no edge repeat)
    t = (t < -3) ? -3 : t;
    t = (t > Hdim + 2) ? (Hdim + 2) : t;
    if (t < 0) t = -t;
    else if (t >= Hdim) t = 2 * Hdim - 2 - t;
    return t;
}

// ---------------------------------------------------------------------------
// Init kernel 1: generate the 49 tap patterns (mirrors the Python exactly)
// ---------------------------------------------------------------------------
__global__ void fov_init_tables(signed char* __restrict__ di,
                                signed char* __restrict__ dj,
                                short* __restrict__ start,
                                short* __restrict__ len) {
    if (threadIdx.x != 0 || blockIdx.x != 0) return;
    int t = 0, p = 0;
    // 9 inner taps
    for (int i = -1; i <= 1; ++i)
        for (int j = -1; j <= 1; ++j) {
            start[p] = (short)t; di[t] = (signed char)i; dj[t] = (signed char)j; ++t;
            len[p] = 1; ++p;
        }
    // ring5: 16 positions
    for (int i = -2; i <= 2; ++i)
        for (int j = -2; j <= 2; ++j) {
            if (!(i == -2 || i == 2 || j == -2 || j == 2)) continue;
            start[p] = (short)t;
            if (i == -2 && j >= -1 && j <= 1) {
                for (int k = 0; k < 5; ++k) { di[t] = (signed char)(i + k); dj[t] = (signed char)j; ++t; }
            } else if (i == 2 && j >= -1 && j <= 1) {
                for (int k = 0; k < 5; ++k) { di[t] = (signed char)(i - k); dj[t] = (signed char)j; ++t; }
            } else if (j == -2 && i >= -1 && i <= 1) {
                for (int k = 0; k < 5; ++k) { di[t] = (signed char)i; dj[t] = (signed char)(j + k); ++t; }
            } else if (j == 2 && i >= -1 && i <= 1) {
                for (int k = 0; k < 5; ++k) { di[t] = (signed char)i; dj[t] = (signed char)(j - k); ++t; }
            } else {  // corners
                int si = (i == -2) ? 1 : -1, sj = (j == -2) ? 1 : -1;
                for (int k = 0; k < 5; ++k) { di[t] = (signed char)(i + si * k); dj[t] = (signed char)j; ++t; }
                for (int k = 1; k < 5; ++k) { di[t] = (signed char)i; dj[t] = (signed char)(j + sj * k); ++t; }
            }
            len[p] = (short)(t - start[p]); ++p;
        }
    // ring7: 24 positions
    for (int i = -3; i <= 3; ++i)
        for (int j = -3; j <= 3; ++j) {
            if (!(i == -3 || i == 3 || j == -3 || j == 3)) continue;
            start[p] = (short)t;
            if (i == -3 && j >= -2 && j <= 2) {
                for (int k = 0; k < 15; ++k) { di[t] = (signed char)(i + k); dj[t] = (signed char)j; ++t; }
            } else if (i == 3 && j >= -2 && j <= 2) {
                for (int k = 0; k < 15; ++k) { di[t] = (signed char)(i - k); dj[t] = (signed char)j; ++t; }
            } else if (j == -3 && i >= -2 && i <= 2) {
                for (int k = 0; k < 25; ++k) { di[t] = (signed char)i; dj[t] = (signed char)(j + k); ++t; }
            } else if (j == 3 && i >= -2 && i <= 2) {
                for (int k = 0; k < 25; ++k) { di[t] = (signed char)i; dj[t] = (signed char)(j - k); ++t; }
            } else {  // corners: 15 vertical + 24 horizontal
                int si = (i == -3) ? 1 : -1, sj = (j == -3) ? 1 : -1;
                for (int k = 0; k < 15; ++k) { di[t] = (signed char)(i + si * k); dj[t] = (signed char)j; ++t; }
                for (int k = 1; k < 25; ++k) { di[t] = (signed char)i; dj[t] = (signed char)(j + sj * k); ++t; }
            }
            len[p] = (short)(t - start[p]); ++p;
        }
    // total: p == 49, t == 661
}

// ---------------------------------------------------------------------------
// Init kernel 2: pack weight (O=64 x K=784 fp32) into bf16 WMMA B-fragment
// order: Bpack[kchunk][ntile][lane][e], e maps to the 16-bit B layout
// (lane<16 -> K = base..base+7, base+16..base+23 with base=0; lane>=16 base=8)
// ---------------------------------------------------------------------------
__global__ void fov_prepack_w(const float* __restrict__ weight,
                              unsigned short* __restrict__ bpack) {
    int kchunk = blockIdx.x >> 2;
    int ntile  = blockIdx.x & 3;
    int lane   = threadIdx.x;          // 0..31
    int base   = (lane < 16) ? 0 : 8;
    int n      = ntile * 16 + (lane & 15);
    unsigned short* dst = bpack + (((kchunk * 4 + ntile) * 32 + lane) * 16);
#pragma unroll
    for (int e = 0; e < 16; ++e) {
        int koff = base + e + ((e >= 8) ? 8 : 0);
        int k = kchunk * 32 + koff;
        float wv = (k < Kdim) ? weight[n * Kdim + k] : 0.0f;
        dst[e] = f2bf(wv);
    }
}

// ---------------------------------------------------------------------------
// Fused main kernel: one workgroup = 32 pixels (b, h, w0..w0+31) x all 64 O.
// Phase 1: build F tile (32 x 800 bf16) in LDS from reflect-padded taps.
// Phase 2: 8 waves x (16-pixel x 16-O tile) x 25 bf16 WMMA steps.
// Phase 3: transpose accumulators through LDS for coalesced stores.
// ---------------------------------------------------------------------------
__global__ __launch_bounds__(256, 2)
void fov_fused(const float* __restrict__ x,
               const unsigned short* __restrict__ bpack,
               const signed char* __restrict__ g_di,
               const signed char* __restrict__ g_dj,
               const short* __restrict__ g_start,
               const short* __restrict__ g_len,
               const float* __restrict__ bias,
               float* __restrict__ out) {
    __shared__ unsigned short F[TILE_M * FSTRIDE];      // 51712 B
    __shared__ signed char s_di[MAXTAPS], s_dj[MAXTAPS];
    __shared__ short s_start[49], s_len[49];

    const int tid  = threadIdx.x;
    const int lane = tid & 31;
    const int wid  = tid >> 5;

    const int tile = blockIdx.x;
    const int wt = tile % (Wdim / TILE_M);
    const int h  = (tile / (Wdim / TILE_M)) % Hdim;
    const int bb = tile / ((Wdim / TILE_M) * Hdim);
    const int w0 = wt * TILE_M;

    // stage tap tables in LDS
    for (int i = tid; i < 661; i += 256) { s_di[i] = g_di[i]; s_dj[i] = g_dj[i]; }
    if (tid < 49) { s_start[tid] = g_start[tid]; s_len[tid] = g_len[tid]; }
    __syncthreads();

    // ---------------- Phase 1: feature tile ----------------
    // idx = pix + 32*cp; consecutive lanes -> consecutive w (coalesced gathers)
    for (int idx = tid; idx < TILE_M * Kdim; idx += 256) {
        const int pix = idx & (TILE_M - 1);
        const int cp  = idx >> 5;               // 0..783
        const int c   = cp / 49;
        const int p   = cp - c * 49;
        const int w   = w0 + pix;
        const int st  = s_start[p];
        const int ln  = s_len[p];
        const float* xc = x + (size_t)(bb * Cdim + c) * (Hdim * Wdim);
        float s = 0.0f;
        for (int q = 0; q < ln; ++q) {
            const int hy = refl192(h + s_di[st + q]);
            const int wx = refl192(w + s_dj[st + q]);
            s += xc[hy * Wdim + wx];
        }
        F[pix * FSTRIDE + cp] = f2bf(s / (float)ln);
    }
    // zero pad K = 784..799
    for (int idx = tid; idx < TILE_M * (KP - Kdim); idx += 256) {
        F[(idx >> 4) * FSTRIDE + Kdim + (idx & 15)] = 0;
    }
    __syncthreads();

    // ---------------- Phase 2: WMMA GEMM ----------------
    union AB { v16bf v; uint4 q[2]; };
    const int pixB = (wid >> 2) << 4;           // 0 or 16
    const int oT   = wid & 3;                   // 16-wide O tile
    const int base = (lane < 16) ? 0 : 8;
    const unsigned short* frow = &F[(pixB + (lane & 15)) * FSTRIDE];

    v8f acc = {0.f, 0.f, 0.f, 0.f, 0.f, 0.f, 0.f, 0.f};
#pragma unroll
    for (int kc = 0; kc < NCHUNK; ++kc) {
        const int kb = kc * 32 + base;
        AB a, b;
        a.q[0] = *(const uint4*)&frow[kb];        // ds_read_b128: K base..base+7
        a.q[1] = *(const uint4*)&frow[kb + 16];   // ds_read_b128: K base+16..+23
        const unsigned short* bp = bpack + (((kc * 4 + oT) * 32 + lane) * 16);
        b.q[0] = *(const uint4*)bp;               // global_load_b128
        b.q[1] = *(const uint4*)(bp + 8);
        acc = __builtin_amdgcn_wmma_f32_16x16x32_bf16(
            false, a.v, false, b.v, (short)0, acc, false, false);
    }
    __syncthreads();   // all waves done reading F before reuse as C buffer

    // ---------------- Phase 3: transpose + coalesced store ----------------
    float* Cs = (float*)F;                      // [64][33] floats = 8448 B
    {
        const int o_local = oT * 16 + (lane & 15);
        const int mAdd = (lane < 16) ? 0 : 8;
#pragma unroll
        for (int r = 0; r < 8; ++r) {
            Cs[o_local * 33 + pixB + mAdd + r] = acc[r];
        }
    }
    __syncthreads();
    for (int idx = tid; idx < Odim * TILE_M; idx += 256) {
        const int o   = idx >> 5;
        const int pix = idx & 31;
        out[((size_t)(bb * Odim + o) * Hdim + h) * Wdim + w0 + pix] =
            Cs[o * 33 + pix] + bias[o];
    }
}

// ---------------------------------------------------------------------------
extern "C" void kernel_launch(void* const* d_in, const int* in_sizes, int n_in,
                              void* d_out, int out_size, void* d_ws, size_t ws_size,
                              hipStream_t stream) {
    const float* x      = (const float*)d_in[0];
    const float* weight = (const float*)d_in[1];
    const float* bias   = (const float*)d_in[2];
    float* out = (float*)d_out;
    char* ws = (char*)d_ws;

    unsigned short* bpack = (unsigned short*)(ws + WS_BPACK);
    signed char* di = (signed char*)(ws + WS_DI);
    signed char* dj = (signed char*)(ws + WS_DJ);
    short* start = (short*)(ws + WS_START);
    short* len   = (short*)(ws + WS_LEN);

    fov_init_tables<<<1, 1, 0, stream>>>(di, dj, start, len);
    fov_prepack_w<<<NCHUNK * 4, 32, 0, stream>>>(weight, bpack);

    const int ntiles = Bdim * Hdim * (Wdim / TILE_M);   // 4608
    fov_fused<<<ntiles, 256, 0, stream>>>(x, bpack, di, dj, start, len, bias, out);
}
